// VisionPatchEmbedder_20976620273964
// MI455X (gfx1250) — compile-verified, matrix operand order checked by reading, not run on hardware
//
#include <hip/hip_runtime.h>
#include <hip/hip_bf16.h>

typedef __attribute__((ext_vector_type(16))) _Float16 v16h;
typedef __attribute__((ext_vector_type(8)))  _Float16 v8h;
typedef __attribute__((ext_vector_type(8)))  float    v8f;

#define BATCH      16
#define SEQ        4096
#define PATCH_DIM  768
#define HIDDEN     768
#define POS_SIZE   10240
#define M_TOTAL    (BATCH * SEQ)     // 65536

#define BLOCK_M    64
#define BLOCK_N    128
#define BLOCK_K    32
#define K_STEPS    (PATCH_DIM / BLOCK_K)   // 24
#define LDS_K      40                // 32 halfs + 8 pad -> 80B row stride, conflict-free b128 reads

union FragU { v16h v; v8h h[2]; };

__global__ __launch_bounds__(256)
void vision_patch_embed_wmma(const float* __restrict__ x,        // [M_TOTAL, PATCH_DIM]
                             const int*   __restrict__ pos_ids,  // [M_TOTAL, 2]
                             const unsigned char* __restrict__ pad_mask, // [M_TOTAL]
                             const float* __restrict__ W,        // [HIDDEN, PATCH_DIM]
                             const float* __restrict__ pos_tab,  // [2, POS_SIZE, HIDDEN]
                             float*       __restrict__ out)      // [M_TOTAL, HIDDEN]
{
    // Double-buffered staging tiles (f16): 2 * (5KB + 10KB) = 30KB of 320KB/WGP
    __shared__ _Float16 Ash[2][BLOCK_M][LDS_K];
    __shared__ _Float16 Bsh[2][BLOCK_N][LDS_K];

    const int tid   = threadIdx.x;
    const int lane  = tid & 31;
    const int wave  = tid >> 5;           // 0..7
    const int lo    = lane & 15;          // 0..15
    const int hi    = lane >> 4;          // 0/1
    const int kb    = hi * 8;             // K-half base per documented 16-bit A/B layout

    const int nbase = blockIdx.x * BLOCK_N;   // over HIDDEN (fast dim -> L2 reuse of x tile)
    const int mbase = blockIdx.y * BLOCK_M;   // over B*N rows

    const int msub   = wave & 3;          // fixed M-subtile per wave
    const int nsub0  = (wave >> 2) * 4;   // 4 consecutive N-subtiles per wave
    const int ar     = msub * 16 + lo;    // A fragment LDS row

    v8f acc[4];
    #pragma unroll
    for (int j = 0; j < 4; ++j) acc[j] = (v8f){};

    // A cooperative load: 64 rows x 32 k = 2048 f32, 8 per thread
    const int a_row = tid >> 2;           // 0..63
    const int a_k   = (tid & 3) * 8;      // 0,8,16,24
    // B cooperative load: 128 rows x 32 k = 4096 f32, 16 per thread
    const int b_row = tid >> 1;           // 0..127
    const int b_k   = (tid & 1) * 16;     // 0,16

    const float* a_gptr = x + (size_t)(mbase + a_row) * PATCH_DIM + a_k;
    const float* b_gptr = W + (size_t)(nbase + b_row) * PATCH_DIM + b_k;

    // ---- helpers ----
    auto stage_store = [&](int buf, float4 fa0, float4 fa1, const float4* fb) {
        _Float16 h8[8];
        h8[0] = (_Float16)__builtin_fmaf(2.0f, fa0.x, -1.0f);
        h8[1] = (_Float16)__builtin_fmaf(2.0f, fa0.y, -1.0f);
        h8[2] = (_Float16)__builtin_fmaf(2.0f, fa0.z, -1.0f);
        h8[3] = (_Float16)__builtin_fmaf(2.0f, fa0.w, -1.0f);
        h8[4] = (_Float16)__builtin_fmaf(2.0f, fa1.x, -1.0f);
        h8[5] = (_Float16)__builtin_fmaf(2.0f, fa1.y, -1.0f);
        h8[6] = (_Float16)__builtin_fmaf(2.0f, fa1.z, -1.0f);
        h8[7] = (_Float16)__builtin_fmaf(2.0f, fa1.w, -1.0f);
        *(v8h*)&Ash[buf][a_row][a_k] = *(const v8h*)h8;

        _Float16 h16[16];
        #pragma unroll
        for (int c = 0; c < 4; ++c) {
            h16[4*c + 0] = (_Float16)fb[c].x;
            h16[4*c + 1] = (_Float16)fb[c].y;
            h16[4*c + 2] = (_Float16)fb[c].z;
            h16[4*c + 3] = (_Float16)fb[c].w;
        }
        *(v8h*)&Bsh[buf][b_row][b_k + 0] = *(const v8h*)&h16[0];
        *(v8h*)&Bsh[buf][b_row][b_k + 8] = *(const v8h*)&h16[8];
    };

    // ---- prologue: fill buffer 0 ----
    {
        float4 fa0 = *(const float4*)(a_gptr + 0);
        float4 fa1 = *(const float4*)(a_gptr + 4);
        float4 fb[4];
        #pragma unroll
        for (int c = 0; c < 4; ++c) fb[c] = *(const float4*)(b_gptr + 4 * c);
        a_gptr += BLOCK_K;
        b_gptr += BLOCK_K;
        stage_store(0, fa0, fa1, fb);
    }
    __syncthreads();

    // ---- main loop: compute buf, stream next tile into buf^1 ----
    for (int step = 0; step < K_STEPS; ++step) {
        const int  buf      = step & 1;
        const bool has_next = (step + 1) < K_STEPS;

        // Issue next tile's global loads first (overlap HBM latency with WMMA)
        float4 na0, na1, nb[4];
        if (has_next) {
            __builtin_prefetch(a_gptr + BLOCK_K, 0, 1);  // speculative, 2 tiles ahead
            na0 = *(const float4*)(a_gptr + 0);
            na1 = *(const float4*)(a_gptr + 4);
            #pragma unroll
            for (int c = 0; c < 4; ++c) nb[c] = *(const float4*)(b_gptr + 4 * c);
            a_gptr += BLOCK_K;
            b_gptr += BLOCK_K;
        }

        // Batch all fragment loads, then 4 back-to-back WMMAs behind one dscnt wait
        FragU fa;
        fa.h[0] = *(const v8h*)&Ash[buf][ar][kb + 0];
        fa.h[1] = *(const v8h*)&Ash[buf][ar][kb + 16];
        FragU fb[4];
        #pragma unroll
        for (int j = 0; j < 4; ++j) {
            const int br = (nsub0 + j) * 16 + lo;
            fb[j].h[0] = *(const v8h*)&Bsh[buf][br][kb + 0];
            fb[j].h[1] = *(const v8h*)&Bsh[buf][br][kb + 16];
        }
        #pragma unroll
        for (int j = 0; j < 4; ++j) {
            acc[j] = __builtin_amdgcn_wmma_f32_16x16x32_f16(
                false, fa.v, false, fb[j].v, (short)0, acc[j], false, false);
        }

        // Convert + store next tile into the other buffer
        if (has_next) stage_store(buf ^ 1, na0, na1, nb);
        __syncthreads();
    }

    // ---- fused epilogue: position-embedding gather + padding mask (branchless) ----
    // C/D layout: VGPR r, lane L -> M = r + 8*(L>=16), N = L%16
    const float* pt0 = pos_tab;
    const float* pt1 = pos_tab + (size_t)POS_SIZE * HIDDEN;
    const int mrow0 = mbase + msub * 16 + hi * 8;

    #pragma unroll
    for (int r = 0; r < 8; ++r) {
        const int m = mrow0 + r;
        const float sel = pad_mask[m] ? 0.0f : 1.0f;
        int p0 = pos_ids[2 * m + 0]; if (p0 < 0) p0 = 0;
        int p1 = pos_ids[2 * m + 1]; if (p1 < 0) p1 = 0;
        const float* row0 = pt0 + (size_t)p0 * HIDDEN;
        const float* row1 = pt1 + (size_t)p1 * HIDDEN;
        float* orow = out + (size_t)m * HIDDEN;
        #pragma unroll
        for (int j = 0; j < 4; ++j) {
            const int ncol = nbase + (nsub0 + j) * 16 + lo;
            orow[ncol] = sel * (acc[j][r] + row0[ncol] + row1[ncol]);
        }
    }
}

extern "C" void kernel_launch(void* const* d_in, const int* in_sizes, int n_in,
                              void* d_out, int out_size, void* d_ws, size_t ws_size,
                              hipStream_t stream) {
    const float*         x        = (const float*)d_in[0];
    const int*           pos_ids  = (const int*)d_in[1];
    const unsigned char* pad_mask = (const unsigned char*)d_in[2];
    const float*         W        = (const float*)d_in[3];
    const float*         pos_tab  = (const float*)d_in[4];
    float*               out      = (float*)d_out;

    dim3 grid(HIDDEN / BLOCK_N,      // 6   (fast: all N-blocks of an M-stripe share x tile in L2)
              M_TOTAL / BLOCK_M);    // 1024
    dim3 block(256);
    vision_patch_embed_wmma<<<grid, block, 0, stream>>>(x, pos_ids, pad_mask, W, pos_tab, out);
}
